// ConvMultiHeadAttention_62027917689195
// MI455X (gfx1250) — matrix operand order repaired and checked by reading, not compiled, not run on hardware
//
#include <hip/hip_runtime.h>
#include <hip/hip_bf16.h>

// ---------------------------------------------------------------------------
// ConvMultiHeadAttention for MI455X (gfx1250, wave32, WMMA bf16)
//   y = softmax((q Wq^T)(k Wk^T)^T / sqrt(D)) (v Wv^T) Wo^T + bo
// All GEMMs + attention on v_wmma_f32_16x16x32_bf16 (f32 accumulate).
// Flash attention (online softmax) avoids the 537MB energy tensor.
// Double-buffered KV staging: K via global_load_async_to_lds_b128
// (ASYNCcnt / s_wait_asynccnt) overlapped with compute; V^T via packed
// b32 LDS transpose with loads issued an iteration early. GEMM fragments
// direct-from-global (bf16 working set ~66MB is L2-resident) + prefetch.
// ---------------------------------------------------------------------------

#define EMBED 512
#define HEADS 8
#define HDIM  64
#define SEQ   1024
#define BATCH 16

typedef __attribute__((ext_vector_type(16))) __bf16 v16bf;
typedef __attribute__((ext_vector_type(8)))  __bf16 v8bf;
typedef __attribute__((ext_vector_type(4)))  __bf16 v4bf;
typedef __attribute__((ext_vector_type(2)))  __bf16 v2bf;
typedef __attribute__((ext_vector_type(8)))  float  v8f;
typedef __attribute__((ext_vector_type(4)))  float  v4f;

// D = A(16x32) * B(32x16) + C, bf16 in / f32 accum
__device__ __forceinline__ v8f wmma_bf16(v16bf a, v16bf b, v8f c) {
  return __builtin_amdgcn_wmma_f32_16x16x32_bf16(
      false, a, false, b, (short)0, c, false, false);
}

__device__ __forceinline__ v16bf combine16(v8bf lo, v8bf hi) {
  v16bf r;
#pragma unroll
  for (int i = 0; i < 8; ++i) { r[i] = lo[i]; r[i + 8] = hi[i]; }
  return r;
}

// A-fragment (16x32, MxK): lane&15 = row M, wave halves split K.
__device__ __forceinline__ v16bf load_frag_a(const __bf16* base, int ld,
                                             int k0, int lane) {
  const __bf16* p = base + (size_t)(lane & 15) * ld + k0 + ((lane >> 4) << 3);
  return combine16(*(const v8bf*)p, *(const v8bf*)(p + 16));
}

// Same fragment but sourced from fp32 with fused cvt to bf16.
__device__ __forceinline__ v16bf load_frag_a_f32(const float* base, int ld,
                                                 int k0, int lane) {
  const float* p = base + (size_t)(lane & 15) * ld + k0 + ((lane >> 4) << 3);
  v4f a0 = *(const v4f*)(p);
  v4f a1 = *(const v4f*)(p + 4);
  v4f a2 = *(const v4f*)(p + 16);
  v4f a3 = *(const v4f*)(p + 20);
  v16bf r;
#pragma unroll
  for (int i = 0; i < 4; ++i) {
    r[i]      = (__bf16)a0[i];
    r[4 + i]  = (__bf16)a1[i];
    r[8 + i]  = (__bf16)a2[i];
    r[12 + i] = (__bf16)a3[i];
  }
  return r;
}

// B-fragment (32x16, KxN) where column n of B is row n of `base` (K-contig):
// lane&15 = column N, wave halves split K (same order as A-fragment).
__device__ __forceinline__ v16bf load_frag_b(const __bf16* base, int ld,
                                             int k0, int lane) {
  const __bf16* p = base + (size_t)(lane & 15) * ld + k0 + ((lane >> 4) << 3);
  return combine16(*(const v8bf*)p, *(const v8bf*)(p + 16));
}

// ---------------------------------------------------------------------------
// fp32 -> bf16 weight conversion (once per weight matrix, 512KB each)
// ---------------------------------------------------------------------------
__global__ void cvt_f32_bf16_kernel(const float* __restrict__ src,
                                    __bf16* __restrict__ dst, int n) {
  int i = (blockIdx.x * blockDim.x + threadIdx.x) * 4;
  if (i + 3 < n) {
    v4f x = *(const v4f*)(src + i);
    v4bf y;
#pragma unroll
    for (int j = 0; j < 4; ++j) y[j] = (__bf16)x[j];
    *(v4bf*)(dst + i) = y;
  }
}

// ---------------------------------------------------------------------------
// Projection GEMM: Y_bf16[M,512] = X_f32[M,512] @ W_bf16[512,512]^T
// Block = 256 thr (8 waves), 128x128 C tile; wave = 32x64 (2x4 WMMA tiles).
// B fragments batched into an array so loads keep distinct registers and
// overlap the WMMA burst (avoid full loadcnt drains).
// ---------------------------------------------------------------------------
__global__ __launch_bounds__(256)
void proj_gemm_kernel(const float* __restrict__ X, const __bf16* __restrict__ W,
                      __bf16* __restrict__ Y) {
  const int lane = threadIdx.x & 31, wid = threadIdx.x >> 5;
  const int m0 = blockIdx.x * 128 + (wid & 3) * 32;
  const int n0 = blockIdx.y * 128 + (wid >> 2) * 64;
  v8f acc[2][4] = {};
  for (int kk = 0; kk < EMBED; kk += 32) {
    // speculative prefetch of next A k-slice (OOB prefetch silently dropped)
    __builtin_prefetch(X + (size_t)(m0 + (lane & 15)) * EMBED + kk + 32, 0, 1);
    v16bf b[4];
#pragma unroll
    for (int ct = 0; ct < 4; ++ct)
      b[ct] = load_frag_b(W + (size_t)(n0 + ct * 16) * EMBED, EMBED, kk, lane);
    v16bf a0 = load_frag_a_f32(X + (size_t)m0 * EMBED, EMBED, kk, lane);
    v16bf a1 = load_frag_a_f32(X + (size_t)(m0 + 16) * EMBED, EMBED, kk, lane);
#pragma unroll
    for (int ct = 0; ct < 4; ++ct) {
      acc[0][ct] = wmma_bf16(a0, b[ct], acc[0][ct]);
      acc[1][ct] = wmma_bf16(a1, b[ct], acc[1][ct]);
    }
  }
  const int hi = lane >> 4, nn = lane & 15;
#pragma unroll
  for (int mi = 0; mi < 2; ++mi)
#pragma unroll
    for (int ct = 0; ct < 4; ++ct)
#pragma unroll
      for (int i = 0; i < 8; ++i) {
        int row = m0 + mi * 16 + hi * 8 + i;
        Y[(size_t)row * EMBED + n0 + ct * 16 + nn] = (__bf16)acc[mi][ct][i];
      }
}

// ---------------------------------------------------------------------------
// Output GEMM: out_f32[M,512] = A_bf16[M,512] @ Wo_bf16[512,512]^T + bo
// ---------------------------------------------------------------------------
__global__ __launch_bounds__(256)
void out_proj_kernel(const __bf16* __restrict__ A, const __bf16* __restrict__ W,
                     const float* __restrict__ bias, float* __restrict__ Y) {
  const int lane = threadIdx.x & 31, wid = threadIdx.x >> 5;
  const int m0 = blockIdx.x * 128 + (wid & 3) * 32;
  const int n0 = blockIdx.y * 128 + (wid >> 2) * 64;
  v8f acc[2][4] = {};
  for (int kk = 0; kk < EMBED; kk += 32) {
    __builtin_prefetch(A + (size_t)(m0 + (lane & 15)) * EMBED + kk + 32, 0, 1);
    v16bf b[4];
#pragma unroll
    for (int ct = 0; ct < 4; ++ct)
      b[ct] = load_frag_b(W + (size_t)(n0 + ct * 16) * EMBED, EMBED, kk, lane);
    v16bf a0 = load_frag_a(A + (size_t)m0 * EMBED, EMBED, kk, lane);
    v16bf a1 = load_frag_a(A + (size_t)(m0 + 16) * EMBED, EMBED, kk, lane);
#pragma unroll
    for (int ct = 0; ct < 4; ++ct) {
      acc[0][ct] = wmma_bf16(a0, b[ct], acc[0][ct]);
      acc[1][ct] = wmma_bf16(a1, b[ct], acc[1][ct]);
    }
  }
  const int hi = lane >> 4, nn = lane & 15;
#pragma unroll
  for (int ct = 0; ct < 4; ++ct) {
    float bv = bias[n0 + ct * 16 + nn];
#pragma unroll
    for (int mi = 0; mi < 2; ++mi)
#pragma unroll
      for (int i = 0; i < 8; ++i) {
        int row = m0 + mi * 16 + hi * 8 + i;
        Y[(size_t)row * EMBED + n0 + ct * 16 + nn] = acc[mi][ct][i] + bv;
      }
  }
}

// ---------------------------------------------------------------------------
// Flash attention per (batch, head). Block = 256 thr (8 waves); each wave
// owns 16 query rows (128 q rows / block); KV swept in 64-row tiles with
// double-buffered staging:
//  - next K tile: global_load_async_to_lds_b128 issued at iteration start,
//    drained with s_wait_asynccnt right before the single per-iter barrier.
//  - next V tile: global loads issued at iteration start, transposed into
//    LDS (packed b32 row-pair stores) after compute, when data has landed.
//  - Q stays in registers for the whole sweep.
//  - P transposed C-layout -> A-layout through a per-wave LDS slice
//    (same-wave DS ops are in-order; no extra barrier needed).
//  - Online softmax stats per-lane (8 rows), shfl_xor over 16-lane halves.
// ---------------------------------------------------------------------------
__global__ __launch_bounds__(256)
void flash_attn_kernel(const __bf16* __restrict__ Qh,
                       const __bf16* __restrict__ Kh,
                       const __bf16* __restrict__ Vh,
                       __bf16* __restrict__ O) {
  __shared__ __bf16 klds[2][64][72];       // K tiles, row-major [kv][d], padded
  __shared__ __bf16 vt[2][64][72];         // V^T tiles [d][kv], padded
  __shared__ __bf16 plds[8][16][HDIM];     // per-wave P tile (16 x 64)

  const int lane = threadIdx.x & 31, wid = threadIdx.x >> 5;
  const int tid = threadIdx.x;
  const int n = blockIdx.z, h = blockIdx.y;
  const int q0 = blockIdx.x * 128 + wid * 16;
  const size_t headoff = (size_t)n * SEQ * EMBED + (size_t)h * HDIM;
  const int hi = lane >> 4, nn = lane & 15;
  const int rp = tid & 31;                 // kv row-pair for V transpose
  const int dc = (tid >> 5) * 8;           // d chunk for V transpose
  const unsigned kbase = (unsigned)(size_t)(&klds[0][0][0]);
  const unsigned KBUFBYTES = 64u * 72u * 2u;   // 9216

  // issue async copy of one K tile (64x64 bf16 = 8KB), 16B per lane per op
  auto issue_k_async = [&](unsigned bufbase, const __bf16* Ksrc) {
#pragma unroll
    for (int j = 0; j < 2; ++j) {
      int c = tid + 256 * j;               // 0..511 chunks of 8 bf16
      int row = c >> 3, d0 = (c & 7) * 8;
      const __bf16* gp = Ksrc + (size_t)row * EMBED + d0;
      unsigned lo = bufbase + (unsigned)(row * 72 + d0) * 2u;
      asm volatile("global_load_async_to_lds_b128 %0, %1, off"
                   :: "v"(lo), "v"(gp) : "memory");
    }
  };

  v16bf qf[2];
  qf[0] = load_frag_a(Qh + headoff + (size_t)q0 * EMBED, EMBED, 0, lane);
  qf[1] = load_frag_a(Qh + headoff + (size_t)q0 * EMBED, EMBED, 32, lane);

  v8f o[4] = {};
  float mrow[8], lrow[8];
#pragma unroll
  for (int i = 0; i < 8; ++i) { mrow[i] = -1e30f; lrow[i] = 0.f; }

  // ---- prologue: stage tile 0 into buffer 0 --------------------------------
  issue_k_async(kbase, Kh + headoff);
  {
    const __bf16* p0 = Vh + headoff + (size_t)(2 * rp) * EMBED + dc;
    v8bf va = *(const v8bf*)p0;
    v8bf vb = *(const v8bf*)(p0 + EMBED);
#pragma unroll
    for (int j = 0; j < 8; ++j) {
      v2bf pk; pk[0] = va[j]; pk[1] = vb[j];
      *(v2bf*)&vt[0][dc + j][2 * rp] = pk;
    }
  }
  asm volatile("s_wait_asynccnt 0x0" ::: "memory");
  __syncthreads();

  for (int kt = 0; kt < SEQ; kt += 64) {
    const int buf = (kt >> 6) & 1;
    const bool has_next = (kt + 64) < SEQ;

    // ---- kick off next tile's staging (overlaps with compute below) ------
    v8bf va, vb;
    if (has_next) {
      issue_k_async(kbase + (unsigned)(buf ^ 1) * KBUFBYTES,
                    Kh + headoff + (size_t)(kt + 64) * EMBED);
      const __bf16* p0 =
          Vh + headoff + (size_t)(kt + 64 + 2 * rp) * EMBED + dc;
      va = *(const v8bf*)p0;
      vb = *(const v8bf*)(p0 + EMBED);
    }

    // ---- S = (Q K^T) / sqrt(D), 16 x 64 per wave (K frags from LDS) ------
    v8f s4[4];
#pragma unroll
    for (int ct = 0; ct < 4; ++ct) {
      v8f s = {};
#pragma unroll
      for (int c = 0; c < 2; ++c) {
        v16bf bk = load_frag_b(&klds[buf][ct * 16][0], 72, 32 * c, lane);
        s = wmma_bf16(qf[c], bk, s);
      }
#pragma unroll
      for (int i = 0; i < 8; ++i) s[i] *= 0.125f;  // 1/sqrt(64)
      s4[ct] = s;
    }

    // ---- online softmax ---------------------------------------------------
    float mnew[8], alpha[8], rs[8];
#pragma unroll
    for (int i = 0; i < 8; ++i) {
      float v0 = fmaxf(fmaxf(s4[0][i], s4[1][i]), fmaxf(s4[2][i], s4[3][i]));
#pragma unroll
      for (int mk = 1; mk < 16; mk <<= 1) v0 = fmaxf(v0, __shfl_xor(v0, mk, 32));
      mnew[i] = fmaxf(mrow[i], v0);
      alpha[i] = __expf(mrow[i] - mnew[i]);
      rs[i] = 0.f;
    }
#pragma unroll
    for (int ct = 0; ct < 4; ++ct)
#pragma unroll
      for (int i = 0; i < 8; ++i) {
        float p = __expf(s4[ct][i] - mnew[i]);
        rs[i] += p;
        o[ct][i] *= alpha[i];
        plds[wid][hi * 8 + i][ct * 16 + nn] = (__bf16)p;  // C-layout -> LDS
      }
#pragma unroll
    for (int i = 0; i < 8; ++i) {
      float r = rs[i];
#pragma unroll
      for (int mk = 1; mk < 16; mk <<= 1) r += __shfl_xor(r, mk, 32);
      lrow[i] = lrow[i] * alpha[i] + r;
      mrow[i] = mnew[i];
    }

    // ---- O += P V (per-wave LDS round-trip; same-wave DS is in-order) ----
#pragma unroll
    for (int c = 0; c < 2; ++c) {
      v16bf ap = load_frag_a(&plds[wid][0][0], HDIM, 32 * c, lane);
#pragma unroll
      for (int ct = 0; ct < 4; ++ct) {
        v16bf bv = load_frag_b(&vt[buf][ct * 16][0], 72, 32 * c, lane);
        o[ct] = wmma_bf16(ap, bv, o[ct]);
      }
    }

    // ---- finish next tile's V^T staging; single barrier per iteration ----
    if (has_next) {
#pragma unroll
      for (int j = 0; j < 8; ++j) {
        v2bf pk; pk[0] = va[j]; pk[1] = vb[j];
        *(v2bf*)&vt[buf ^ 1][dc + j][2 * rp] = pk;
      }
    }
    asm volatile("s_wait_asynccnt 0x0" ::: "memory");
    __syncthreads();
  }

  // ---- normalize and write attn output (bf16, [N,L,H,D] layout) ----------
#pragma unroll
  for (int ct = 0; ct < 4; ++ct)
#pragma unroll
    for (int i = 0; i < 8; ++i) {
      int row = q0 + hi * 8 + i;
      O[headoff + (size_t)row * EMBED + ct * 16 + nn] =
          (__bf16)(o[ct][i] / lrow[i]);
    }
}

// ---------------------------------------------------------------------------
extern "C" void kernel_launch(void* const* d_in, const int* in_sizes, int n_in,
                              void* d_out, int out_size, void* d_ws,
                              size_t ws_size, hipStream_t stream) {
  const float* v  = (const float*)d_in[0];
  const float* k  = (const float*)d_in[1];
  const float* q  = (const float*)d_in[2];
  const float* Wv = (const float*)d_in[3];
  const float* Wk = (const float*)d_in[4];
  const float* Wq = (const float*)d_in[5];
  const float* Wo = (const float*)d_in[6];
  const float* bo = (const float*)d_in[7];

  const size_t WELEMS = (size_t)EMBED * EMBED;        // 262144
  const size_t XELEMS = (size_t)BATCH * SEQ * EMBED;  // 8388608
  __bf16* ws   = (__bf16*)d_ws;                       // ~66MB bf16 total
  __bf16* Wq_b = ws;
  __bf16* Wk_b = Wq_b + WELEMS;
  __bf16* Wv_b = Wk_b + WELEMS;
  __bf16* Wo_b = Wv_b + WELEMS;
  __bf16* qh   = Wo_b + WELEMS;
  __bf16* kh   = qh + XELEMS;
  __bf16* vh   = kh + XELEMS;
  __bf16* ao   = vh + XELEMS;

  // weights fp32 -> bf16 (tiny)
  cvt_f32_bf16_kernel<<<256, 256, 0, stream>>>(Wq, Wq_b, (int)WELEMS);
  cvt_f32_bf16_kernel<<<256, 256, 0, stream>>>(Wk, Wk_b, (int)WELEMS);
  cvt_f32_bf16_kernel<<<256, 256, 0, stream>>>(Wv, Wv_b, (int)WELEMS);
  cvt_f32_bf16_kernel<<<256, 256, 0, stream>>>(Wo, Wo_b, (int)WELEMS);

  // q/k/v projections (fused fp32->bf16 A-fragment loads)
  dim3 gproj(BATCH * SEQ / 128, EMBED / 128);  // (128, 4)
  proj_gemm_kernel<<<gproj, 256, 0, stream>>>(q, Wq_b, qh);
  proj_gemm_kernel<<<gproj, 256, 0, stream>>>(k, Wk_b, kh);
  proj_gemm_kernel<<<gproj, 256, 0, stream>>>(v, Wv_b, vh);

  // flash attention: grid (q-tiles, heads, batch)
  flash_attn_kernel<<<dim3(SEQ / 128, HEADS, BATCH), 256, 0, stream>>>(
      qh, kh, vh, ao);

  // output projection + bias -> fp32 d_out
  out_proj_kernel<<<gproj, 256, 0, stream>>>(ao, Wo_b, bo, (float*)d_out);
}
